// Model_44169443672642
// MI455X (gfx1250) — matrix operand order repaired and checked by reading, not compiled, not run on hardware
//
#include <hip/hip_runtime.h>
#include <hip/hip_bf16.h>
#include <stdint.h>

// ---------------------------------------------------------------------------
// Types / WMMA helpers (gfx1250, wave32)
// ---------------------------------------------------------------------------
typedef __attribute__((ext_vector_type(16))) _Float16 v16h;
typedef __attribute__((ext_vector_type(8)))  _Float16 v8h;
typedef __attribute__((ext_vector_type(8)))  float    v8f;

#define DEV static __device__ __forceinline__

DEV v16h cat16(v8h lo, v8h hi) {
  return __builtin_shufflevector(lo, hi, 0,1,2,3,4,5,6,7,8,9,10,11,12,13,14,15);
}

// A fragment (16x32 f16): lane -> row m0+(lane&15); halves 0..7 -> k0+koff+0..7,
// halves 8..15 -> k0+koff+16..23 where koff = 8 for lanes 16..31.
DEV v16h load_a_frag(const _Float16* A, int lda, int m0, int k0, int lane) {
  const _Float16* p = A + (size_t)(m0 + (lane & 15)) * lda + (k0 + ((lane & 16) ? 8 : 0));
  v8h lo = *(const v8h*)p;
  v8h hi = *(const v8h*)(p + 16);
  return cat16(lo, hi);
}

// B fragment (32x16 f16): lane -> column n0+(lane&15) of K-major weight Wt[K][ldb];
// halves 0..7 -> rows k0+koff+0..7, halves 8..15 -> rows k0+koff+16..23, koff = 8 for lanes 16..31.
DEV v16h load_b_frag(const _Float16* Wt, int ldb, int k0, int n0, int lane) {
  const _Float16* p = Wt + (size_t)(k0 + ((lane & 16) ? 8 : 0)) * ldb + n0 + (lane & 15);
  v16h b;
  #pragma unroll
  for (int s = 0; s < 8; ++s) {
    b[s]     = p[(size_t)s * ldb];
    b[8 + s] = p[(size_t)(16 + s) * ldb];
  }
  return b;
}

DEV v8f wmma16(v16h a, v16h b, v8f c) {
  return __builtin_amdgcn_wmma_f32_16x16x32_f16(false, a, false, b, (short)0, c, false, false);
}

// One 16x16 output tile, K multiple of 32 (compile-time constant at call sites).
DEV v8f tile_mm(const _Float16* A, int lda, const _Float16* Wt, int ldb,
                int m0, int n0, int K, int lane) {
  v8f acc = {};
  #pragma unroll
  for (int k0 = 0; k0 < K; k0 += 32) {
    v16h a = load_a_frag(A, lda, m0, k0, lane);
    v16h b = load_b_frag(Wt, ldb, k0, n0, lane);
    acc = wmma16(a, b, acc);
  }
  return acc;
}

// ---------------------------------------------------------------------------
// Generic GEMM: C[M,N] = A[M,K](f16) @ W[N,K]^T + bias
// Register-blocked 4 N-tiles per wave: one A fragment feeds 4 back-to-back
// WMMAs per K-step (16 v_wmma total, fully unrolled, branch-free MAC loop).
// Wt is K-major with leading dim ldb padded to a multiple of 64 columns.
// M must be a multiple of 16; stores guarded on N only.
// ---------------------------------------------------------------------------
template <int K>
__global__ __launch_bounds__(256) void gemm_f16(
    const _Float16* __restrict__ A, const _Float16* __restrict__ Wt,
    const float* __restrict__ bias, float* __restrict__ C,
    int M, int N, int ldb, int ldc)
{
  const int lane = threadIdx.x & 31;
  const int wv   = threadIdx.x >> 5;
  const int MT = M >> 4;
  const int NT = (N + 15) >> 4;
  const int NG = (NT + 3) >> 2;          // groups of 4 N-tiles
  int t = blockIdx.x * 8 + wv;
  if (t >= MT * NG) return;
  const int mt = t / NG, ng = t % NG;
  const int m0 = mt * 16, n0 = ng * 64;
  v8f accs[4];
  #pragma unroll
  for (int j = 0; j < 4; ++j) accs[j] = (v8f){};
  #pragma unroll
  for (int k0 = 0; k0 < K; k0 += 32) {
    v16h a = load_a_frag(A, K, m0, k0, lane);
    #pragma unroll
    for (int j = 0; j < 4; ++j)
      accs[j] = wmma16(a, load_b_frag(Wt, ldb, k0, n0 + j * 16, lane), accs[j]);
  }
  #pragma unroll
  for (int j = 0; j < 4; ++j) {
    const int nrow = n0 + j * 16 + (lane & 15);
    if (nrow < N) {
      float bv = bias ? bias[nrow] : 0.f;
      #pragma unroll
      for (int r = 0; r < 8; ++r) {
        int m = m0 + r + ((lane & 16) ? 8 : 0);
        C[(size_t)m * ldc + nrow] = accs[j][r] + bv;
      }
    }
  }
}

// ---------------------------------------------------------------------------
// Small glue kernels
// ---------------------------------------------------------------------------
__global__ void fill_u32_k(uint32_t* p, uint32_t v, int n) {
  int i = blockIdx.x * 256 + threadIdx.x;
  if (i < n) p[i] = v;
}

// W[N][K] f32 -> Wt[K][ldb] f16 (K-major, padded columns pre-zeroed)
__global__ void transpose_f16_k(const float* __restrict__ W, _Float16* __restrict__ Wt,
                                int N, int K, int ldb) {
  int i = blockIdx.x * 256 + threadIdx.x;
  if (i < N * K) {
    int n = i / K, k = i % K;
    Wt[(size_t)k * ldb + n] = (_Float16)W[i];
  }
}

// H0 = LN(gelu(static @ seg_fc_w^T + b))   (K=16, pure VALU — negligible work)
__global__ __launch_bounds__(128) void h0_kernel(
    const float* __restrict__ st, const float* __restrict__ W, const float* __restrict__ b,
    const float* __restrict__ g, const float* __restrict__ bb,
    float* __restrict__ H0, _Float16* __restrict__ H0h)
{
  __shared__ float feat[16];
  __shared__ float val[128];
  const int node = blockIdx.x, d = threadIdx.x;
  if (d < 16) feat[d] = st[node * 16 + d];
  __syncthreads();
  float acc = b[d];
  #pragma unroll
  for (int k = 0; k < 16; ++k) acc += feat[k] * W[d * 16 + k];
  float x = acc;
  float c = x + 0.044715f * x * x * x;
  float ge = 0.5f * x * (1.f + tanhf(0.7978845608028654f * c));
  val[d] = ge;
  __syncthreads();
  float mu = 0.f;
  for (int k = 0; k < 128; ++k) mu += val[k];
  mu *= (1.f / 128.f);
  float var = 0.f;
  for (int k = 0; k < 128; ++k) { float t = val[k] - mu; var += t * t; }
  var *= (1.f / 128.f);
  float out = (ge - mu) * rsqrtf(var + 1e-5f) * g[d] + bb[d];
  H0 [(size_t)node * 128 + d] = out;
  H0h[(size_t)node * 128 + d] = (_Float16)out;
}

// per-node dot products with attention vectors
__global__ void gat_dots_k(const float* __restrict__ q, const float* __restrict__ k,
                           const float* __restrict__ asrc, const float* __restrict__ adst,
                           float* __restrict__ ka, float* __restrict__ qa, int n)
{
  int i = blockIdx.x * 256 + threadIdx.x;
  if (i < n) {
    float a = 0.f, b = 0.f;
    for (int d = 0; d < 128; ++d) {
      a += k[(size_t)i * 128 + d] * asrc[d];
      b += q[(size_t)i * 128 + d] * adst[d];
    }
    ka[i] = a; qa[i] = b;
  }
}

DEV unsigned fkey(float f) {
  unsigned u = __float_as_uint(f);
  return (u & 0x80000000u) ? ~u : (u | 0x80000000u);
}
DEV float unkey(unsigned kk) {
  unsigned u = (kk & 0x80000000u) ? (kk & 0x7FFFFFFFu) : ~kk;
  return __uint_as_float(u);
}

__global__ void gat_logits_k(const int* __restrict__ src, const int* __restrict__ dst,
                             const float* __restrict__ ka, const float* __restrict__ qa,
                             const float* __restrict__ st, int use_deter,
                             float* __restrict__ logits, unsigned* __restrict__ segmax, int E)
{
  int e = blockIdx.x * 256 + threadIdx.x;
  if (e < E) {
    int s = src[e], d = dst[e];
    float l = ka[s] + qa[d];
    l = (l > 0.f) ? l : 0.2f * l;                    // leaky_relu(., 0.2)
    if (use_deter) {
      float dx = st[(size_t)s * 16 + 0] - st[(size_t)d * 16 + 0];
      float dy = st[(size_t)s * 16 + 1] - st[(size_t)d * 16 + 1];
      l += -0.5f * (dx * dx + dy * dy + 1e-9f);      // -d_e^2/2
    }
    logits[e] = l;
    atomicMax(&segmax[d], fkey(l));
  }
}

__global__ void gat_expsum_k(const int* __restrict__ dst, const unsigned* __restrict__ segmax,
                             float* __restrict__ logits, float* __restrict__ segsum, int E)
{
  int e = blockIdx.x * 256 + threadIdx.x;
  if (e < E) {
    int d = dst[e];
    float ex = __expf(logits[e] - unkey(segmax[d]));
    logits[e] = ex;
    atomicAdd(&segsum[d], ex);
  }
}

__global__ void gat_agg_k(const int* __restrict__ src, const int* __restrict__ dst,
                          const float* __restrict__ eexp, const float* __restrict__ segsum,
                          const float* __restrict__ k, float* __restrict__ out, int E)
{
  int idx = blockIdx.x * 256 + threadIdx.x;
  int e = idx >> 5, lane = idx & 31;
  if (e < E) {
    float attn = eexp[e] / (segsum[dst[e]] + 1e-9f);
    const float* kr = k + (size_t)src[e] * 128;
    float* orow = out + (size_t)dst[e] * 128;
    for (int c = lane; c < 128; c += 32) atomicAdd(&orow[c], attn * kr[c]);
  }
}

// X_tt[node][t][d] = S_hist@tt_in_w^T + b + weekly + daily + pos + X_G
__global__ __launch_bounds__(128) void build_tt_k(
    const float* __restrict__ S_hist, const float* __restrict__ w, const float* __restrict__ b,
    const float* __restrict__ weekly, const float* __restrict__ daily, const float* __restrict__ posb,
    const int* __restrict__ widx, const int* __restrict__ didx,
    const float* __restrict__ XG, float* __restrict__ X, int n)
{
  int node = blockIdx.x, t = blockIdx.y, d = threadIdx.x;
  float s0 = S_hist[((size_t)t * n + node) * 2 + 0];
  float s1 = S_hist[((size_t)t * n + node) * 2 + 1];
  float v = s0 * w[d * 2 + 0] + s1 * w[d * 2 + 1] + b[d]
          + weekly[widx[t] * 128 + d] + daily[didx[t] * 128 + d] + posb[t * 128 + d]
          + XG[(size_t)node * 128 + d];
  X[((size_t)node * 24 + t) * 128 + d] = v;
}

__global__ __launch_bounds__(256) void last_tok_k(const float* __restrict__ X,
                                                  _Float16* __restrict__ out, int n) {
  int i = blockIdx.x * 256 + threadIdx.x;
  if (i < 5008 * 128) {
    int node = i >> 7, d = i & 127;
    out[i] = (node < n) ? (_Float16)X[((size_t)node * 24 + 23) * 128 + d] : (_Float16)0.f;
  }
}

// H = H0 + H_traf_time + H_traj, gathered by traj_nodes, to f16
__global__ __launch_bounds__(128) void gather_h_k(
    const float* __restrict__ H0, const float* __restrict__ Htt, const float* __restrict__ Htraj,
    const int* __restrict__ nodes, _Float16* __restrict__ out)
{
  int i = blockIdx.x, d = threadIdx.x;
  size_t o = (size_t)nodes[i] * 128 + d;
  out[(size_t)i * 128 + d] = (_Float16)(H0[o] + Htt[o] + Htraj[o]);
}

__global__ __launch_bounds__(128) void build_xt_k(
    const float* __restrict__ G, const float* __restrict__ times,
    const float* __restrict__ tw, const float* __restrict__ tb,
    const float* __restrict__ cls, float* __restrict__ xt)
{
  int bidx = blockIdx.x, srow = blockIdx.y, d = threadIdx.x;
  float v;
  if (srow == 0) v = cls[d];
  else {
    int i = bidx * 64 + (srow - 1);
    v = G[(size_t)i * 128 + d] + times[i] * tw[d] + tb[d];
  }
  xt[((size_t)bidx * 65 + srow) * 128 + d] = v;
}

__global__ __launch_bounds__(128) void pos_gather_k(const float* __restrict__ xt,
                                                    _Float16* __restrict__ posh) {
  int i = blockIdx.x, d = threadIdx.x;
  int b = i >> 6, l = i & 63;
  posh[(size_t)i * 128 + d] = (_Float16)xt[((size_t)b * 65 + 1 + l) * 128 + d];
}

// ---------------------------------------------------------------------------
// Fused transformer encoder layer (one sequence per workgroup, 8 waves)
// S = real sequence length, SP = padded (mult of 16). D=128, NH=8, dh=16.
// LDS: xf (f32 residual), xh (f16 A-operand), qk (f16 qkv / ffn hidden).
// ---------------------------------------------------------------------------
DEV void ln_row(float* row, const float* g, const float* b,
                _Float16* out16, float* gout) {
  float mu = 0.f;
  for (int d = 0; d < 128; ++d) mu += row[d];
  mu *= (1.f / 128.f);
  float var = 0.f;
  for (int d = 0; d < 128; ++d) { float t = row[d] - mu; var += t * t; }
  var *= (1.f / 128.f);
  float inv = rsqrtf(var + 1e-5f);
  for (int d = 0; d < 128; ++d) {
    float v = (row[d] - mu) * inv * g[d] + b[d];
    row[d] = v;
    if (out16) out16[d] = (_Float16)v;
    if (gout)  gout[d] = v;
  }
}

template <int S, int SP>
__global__ __launch_bounds__(256) void enc_layer_kernel(
    float* __restrict__ X,
    const _Float16* __restrict__ Wqkv_t, const float* __restrict__ bqkv,
    const _Float16* __restrict__ Wo_t,   const float* __restrict__ bo,
    const float* __restrict__ ln1g, const float* __restrict__ ln1b,
    const _Float16* __restrict__ W1_t, const float* __restrict__ b1,
    const _Float16* __restrict__ W2_t, const float* __restrict__ b2,
    const float* __restrict__ ln2g, const float* __restrict__ ln2b)
{
  __shared__ float    xf[SP][132];
  __shared__ _Float16 xh[SP][136];
  __shared__ _Float16 qk[SP][392];

  const int tid = threadIdx.x;
  const int lane = tid & 31;
  const int wv = tid >> 5;
  float* Xs = X + (size_t)blockIdx.x * S * 128;

  for (int i = tid; i < SP * 128; i += 256) {
    int m = i >> 7, d = i & 127;
    float v = (m < S) ? Xs[m * 128 + d] : 0.f;
    xf[m][d] = v;
    xh[m][d] = (_Float16)v;
  }
  __syncthreads();

  // ---- QKV: qk[SP][384] = xh @ Wqkv^T + bqkv (f16) ----
  for (int t = wv; t < (SP / 16) * 24; t += 8) {
    int mt = t / 24, nt = t % 24;
    v8f acc = tile_mm(&xh[0][0], 136, Wqkv_t, 384, mt * 16, nt * 16, 128, lane);
    int n = nt * 16 + (lane & 15);
    float bv = bqkv[n];
    #pragma unroll
    for (int r = 0; r < 8; ++r) {
      int m = mt * 16 + r + ((lane & 16) ? 8 : 0);
      qk[m][n] = (_Float16)(acc[r] + bv);
    }
  }
  __syncthreads();

  // ---- Attention (VALU flash-style; dh=16 tiny) -> xh ----
  for (int pid = tid; pid < S * 8; pid += 256) {
    const int h = pid & 7, m = pid >> 3;
    const int qb = h * 16, kb = 128 + h * 16, vb = 256 + h * 16;
    float qv[16], oacc[16];
    #pragma unroll
    for (int j = 0; j < 16; ++j) { qv[j] = (float)qk[m][qb + j]; oacc[j] = 0.f; }
    float mx = -3.0e38f, den = 0.f;
    for (int s2 = 0; s2 < S; ++s2) {
      float dt = 0.f;
      #pragma unroll
      for (int j = 0; j < 16; ++j) dt += qv[j] * (float)qk[s2][kb + j];
      dt *= 0.25f;                       // 1/sqrt(16)
      float nm = fmaxf(mx, dt);
      float sc = __expf(mx - nm);
      float pw = __expf(dt - nm);
      den = den * sc + pw;
      #pragma unroll
      for (int j = 0; j < 16; ++j) oacc[j] = oacc[j] * sc + pw * (float)qk[s2][vb + j];
      mx = nm;
    }
    float inv = 1.f / den;
    #pragma unroll
    for (int j = 0; j < 16; ++j) xh[m][qb + j] = (_Float16)(oacc[j] * inv);
  }
  __syncthreads();

  // ---- proj + residual into xf ----
  for (int t = wv; t < (SP / 16) * 8; t += 8) {
    int mt = t / 8, nt = t % 8;
    v8f acc = tile_mm(&xh[0][0], 136, Wo_t, 128, mt * 16, nt * 16, 128, lane);
    int n = nt * 16 + (lane & 15);
    float bv = bo[n];
    #pragma unroll
    for (int r = 0; r < 8; ++r) {
      int m = mt * 16 + r + ((lane & 16) ? 8 : 0);
      xf[m][n] += acc[r] + bv;
    }
  }
  __syncthreads();

  if (tid < SP) ln_row(xf[tid], ln1g, ln1b, xh[tid], nullptr);
  __syncthreads();

  // ---- FFN1: qk[SP][256] = relu(xh @ W1^T + b1) (f16) ----
  for (int t = wv; t < (SP / 16) * 16; t += 8) {
    int mt = t / 16, nt = t % 16;
    v8f acc = tile_mm(&xh[0][0], 136, W1_t, 256, mt * 16, nt * 16, 128, lane);
    int n = nt * 16 + (lane & 15);
    float bv = b1[n];
    #pragma unroll
    for (int r = 0; r < 8; ++r) {
      int m = mt * 16 + r + ((lane & 16) ? 8 : 0);
      qk[m][n] = (_Float16)fmaxf(acc[r] + bv, 0.f);
    }
  }
  __syncthreads();

  // ---- FFN2 + residual ----
  for (int t = wv; t < (SP / 16) * 8; t += 8) {
    int mt = t / 8, nt = t % 8;
    v8f acc = tile_mm(&qk[0][0], 392, W2_t, 128, mt * 16, nt * 16, 256, lane);
    int n = nt * 16 + (lane & 15);
    float bv = b2[n];
    #pragma unroll
    for (int r = 0; r < 8; ++r) {
      int m = mt * 16 + r + ((lane & 16) ? 8 : 0);
      xf[m][n] += acc[r] + bv;
    }
  }
  __syncthreads();

  if (tid < SP) ln_row(xf[tid], ln2g, ln2b, nullptr, (tid < S) ? (Xs + tid * 128) : nullptr);
}

// ---------------------------------------------------------------------------
// Parameter layout resolution (handles leaf-per-entry insertion / sorted,
// or one concatenated sorted blob).
// ---------------------------------------------------------------------------
#define ENC_ENUM(p) p##_Wqkv, p##_bqkv, p##_Wo, p##_bo, p##_ln1g, p##_ln1b, \
                    p##_W1, p##_b1, p##_W2, p##_b2, p##_ln2g, p##_ln2b
enum PIdx {
  P_seg_fc_w = 0, P_seg_fc_b, P_seg_ln_g, P_seg_ln_b,
  P_tg_W, P_tg_asrc, P_tg_adst,
  P_traf_Wq, P_traf_Wk, P_traf_asrc, P_traf_adst,
  P_tt_in_w, P_tt_in_b, P_tt_weekly, P_tt_daily, P_tt_pos,
  ENC_ENUM(P_tt0), ENC_ENUM(P_tt1),
  P_tt_out_w, P_tt_out_b,
  P_tj_node_w, P_tj_node_b, P_tj_time_w, P_tj_time_b, P_tj_cls,
  ENC_ENUM(P_tj0), ENC_ENUM(P_tj1),
  P_mtp_seg_w, P_mtp_seg_b,
  NPARAM
};
#define ENC_SZ 49152, 384, 16384, 128, 128, 128, 32768, 256, 32768, 128, 128, 128
static const int PSZ[NPARAM] = {
  2048, 128, 128, 128,
  16384, 128, 128,
  16384, 16384, 128, 128,
  256, 128, 896, 3072, 3328,
  ENC_SZ, ENC_SZ,
  16384, 128,
  16384, 128, 128, 128, 128,
  ENC_SZ, ENC_SZ,
  640000, 5000
};
#define ENC_SORT(p) p##_W1, p##_W2, p##_Wo, p##_Wqkv, p##_b1, p##_b2, p##_bo, p##_bqkv, \
                    p##_ln1b, p##_ln1g, p##_ln2b, p##_ln2g
static const int SORTED[NPARAM] = {
  P_mtp_seg_b, P_mtp_seg_w, P_seg_fc_b, P_seg_fc_w, P_seg_ln_b, P_seg_ln_g,
  P_tg_W, P_tg_adst, P_tg_asrc, P_tj_cls,
  ENC_SORT(P_tj0), ENC_SORT(P_tj1),
  P_tj_node_b, P_tj_node_w, P_tj_time_b, P_tj_time_w,
  P_traf_Wk, P_traf_Wq, P_traf_adst, P_traf_asrc,
  P_tt_daily, P_tt_in_b, P_tt_in_w,
  ENC_SORT(P_tt0), ENC_SORT(P_tt1),
  P_tt_out_b, P_tt_out_w, P_tt_pos, P_tt_weekly
};

// ---------------------------------------------------------------------------
// Host orchestration
// ---------------------------------------------------------------------------
extern "C" void kernel_launch(void* const* d_in, const int* in_sizes, int n_in,
                              void* d_out, int out_size, void* d_ws, size_t ws_size,
                              hipStream_t stream)
{
  (void)out_size; (void)ws_size;
  const int NN = 5000, NNP = 5008, EE = 80000, BL = 4096;
  const int MTP_LD = 5056;              // 79 * 64, N-group aligned

  const float* statc  = (const float*)d_in[0];
  const float* S_hist = (const float*)d_in[1];
  const float* times  = (const float*)d_in[2];
  const int* edges  = (const int*)d_in[n_in - 4];
  const int* tnodes = (const int*)d_in[n_in - 3];
  const int* widx   = (const int*)d_in[n_in - 2];
  const int* didx   = (const int*)d_in[n_in - 1];
  const int* esrc = edges;
  const int* edst = edges + EE;

  const float* prm[NPARAM];
  if (n_in >= 3 + NPARAM + 4) {
    if (in_sizes[3] == 2048) {        // insertion order, leaf per entry
      for (int i = 0; i < NPARAM; ++i) prm[i] = (const float*)d_in[3 + i];
    } else {                          // jax tree_flatten (sorted) order
      for (int j = 0; j < NPARAM; ++j) prm[SORTED[j]] = (const float*)d_in[3 + j];
    }
  } else {                            // params concatenated into one blob (sorted)
    const float* base = (const float*)d_in[3];
    size_t off = 0;
    for (int j = 0; j < NPARAM; ++j) { prm[SORTED[j]] = base + off; off += PSZ[SORTED[j]]; }
  }

  // ---- workspace bump allocator ----
  char* wp = (char*)d_ws;
  auto alloc = [&](size_t bytes) -> void* {
    void* p = (void*)wp; wp += (bytes + 255) & ~(size_t)255; return p;
  };
  _Float16* t_tgW    = (_Float16*)alloc(16384 * 2);
  _Float16* t_trafWq = (_Float16*)alloc(16384 * 2);
  _Float16* t_trafWk = (_Float16*)alloc(16384 * 2);
  _Float16* t_ttout  = (_Float16*)alloc(16384 * 2);
  _Float16* t_tjnode = (_Float16*)alloc(16384 * 2);
  _Float16* t_mtp    = (_Float16*)alloc((size_t)128 * MTP_LD * 2);  // padded cols
  _Float16* t_enc[4][4];                      // [tt0,tt1,tj0,tj1][Wqkv,Wo,W1,W2]
  static const int enc_wsz[4] = {49152, 16384, 32768, 32768};
  for (int l = 0; l < 4; ++l)
    for (int m = 0; m < 4; ++m) t_enc[l][m] = (_Float16*)alloc((size_t)enc_wsz[m] * 2);

  float*    H0    = (float*)alloc((size_t)NNP * 128 * 4);
  _Float16* H0h   = (_Float16*)alloc((size_t)NNP * 128 * 2);
  float*    qb    = (float*)alloc((size_t)NNP * 128 * 4);
  float*    kb    = (float*)alloc((size_t)NNP * 128 * 4);
  float*    XG    = (float*)alloc((size_t)NNP * 128 * 4);
  float*    Htraj = (float*)alloc((size_t)NNP * 128 * 4);
  float*    Htt   = (float*)alloc((size_t)NNP * 128 * 4);
  float*    ka    = (float*)alloc((size_t)NNP * 4);
  float*    qa    = (float*)alloc((size_t)NNP * 4);
  unsigned* smax  = (unsigned*)alloc((size_t)NNP * 4);
  float*    ssum  = (float*)alloc((size_t)NNP * 4);
  float*    elog  = (float*)alloc((size_t)EE * 4);
  float*    Xtt   = (float*)alloc((size_t)NN * 24 * 128 * 4);
  _Float16* xlh   = (_Float16*)alloc((size_t)NNP * 128 * 2);
  _Float16* Hgh   = (_Float16*)alloc((size_t)BL * 128 * 2);
  float*    Gb    = (float*)alloc((size_t)BL * 128 * 4);
  float*    xt    = (float*)alloc((size_t)64 * 65 * 128 * 4);
  _Float16* posh  = (_Float16*)alloc((size_t)BL * 128 * 2);

  auto fill = [&](void* p, uint32_t v, int n) {
    fill_u32_k<<<(n + 255) / 256, 256, 0, stream>>>((uint32_t*)p, v, n);
  };
  auto tr = [&](int pid, _Float16* wt, int N, int K, int ldb) {
    int n = N * K;
    transpose_f16_k<<<(n + 255) / 256, 256, 0, stream>>>(prm[pid], wt, N, K, ldb);
  };
  auto gemm = [&](const _Float16* A, const _Float16* Wt, const float* bias,
                  float* C, int M, int N, int ldb, int ldc) {
    int NT = (N + 15) / 16;
    int tiles = (M / 16) * ((NT + 3) / 4);
    gemm_f16<128><<<(tiles + 7) / 8, 256, 0, stream>>>(A, Wt, bias, C, M, N, ldb, ldc);
  };

  // ---- weight conversion (f32 -> K-major f16; mtp padded & pre-zeroed) ----
  tr(P_tg_W, t_tgW, 128, 128, 128);
  tr(P_traf_Wq, t_trafWq, 128, 128, 128);
  tr(P_traf_Wk, t_trafWk, 128, 128, 128);
  tr(P_tt_out_w, t_ttout, 128, 128, 128);
  tr(P_tj_node_w, t_tjnode, 128, 128, 128);
  fill(t_mtp, 0u, 128 * MTP_LD / 2);
  tr(P_mtp_seg_w, t_mtp, 5000, 128, MTP_LD);
  tr(P_tt0_Wqkv, t_enc[0][0], 384, 128, 384); tr(P_tt0_Wo, t_enc[0][1], 128, 128, 128);
  tr(P_tt0_W1,   t_enc[0][2], 256, 128, 256); tr(P_tt0_W2, t_enc[0][3], 128, 256, 128);
  tr(P_tt1_Wqkv, t_enc[1][0], 384, 128, 384); tr(P_tt1_Wo, t_enc[1][1], 128, 128, 128);
  tr(P_tt1_W1,   t_enc[1][2], 256, 128, 256); tr(P_tt1_W2, t_enc[1][3], 128, 256, 128);
  tr(P_tj0_Wqkv, t_enc[2][0], 384, 128, 384); tr(P_tj0_Wo, t_enc[2][1], 128, 128, 128);
  tr(P_tj0_W1,   t_enc[2][2], 256, 128, 256); tr(P_tj0_W2, t_enc[2][3], 128, 256, 128);
  tr(P_tj1_Wqkv, t_enc[3][0], 384, 128, 384); tr(P_tj1_Wo, t_enc[3][1], 128, 128, 128);
  tr(P_tj1_W1,   t_enc[3][2], 256, 128, 256); tr(P_tj1_W2, t_enc[3][3], 128, 256, 128);

  // ---- H0 ----
  fill(H0h, 0u, NNP * 128 / 2);   // zero f16 buffer (incl. pad rows), 2 halves/dword
  h0_kernel<<<NN, 128, 0, stream>>>(statc, prm[P_seg_fc_w], prm[P_seg_fc_b],
                                    prm[P_seg_ln_g], prm[P_seg_ln_b], H0, H0h);

  // ---- GAT #1 (traffic, with deterrence) -> X_G ----
  gemm(H0h, t_trafWq, nullptr, qb, NNP, 128, 128, 128);
  gemm(H0h, t_trafWk, nullptr, kb, NNP, 128, 128, 128);
  gat_dots_k<<<(NN + 255) / 256, 256, 0, stream>>>(qb, kb, prm[P_traf_asrc], prm[P_traf_adst], ka, qa, NN);
  fill(smax, 0x00800000u, NNP);            // key(-FLT_MAX)
  fill(ssum, 0u, NNP);
  fill(XG, 0u, NNP * 128);
  gat_logits_k<<<(EE + 255) / 256, 256, 0, stream>>>(esrc, edst, ka, qa, statc, 1, elog, smax, EE);
  gat_expsum_k<<<(EE + 255) / 256, 256, 0, stream>>>(edst, smax, elog, ssum, EE);
  gat_agg_k<<<(EE * 32 + 255) / 256, 256, 0, stream>>>(esrc, edst, elog, ssum, kb, XG, EE);

  // ---- temporal transformer input ----
  build_tt_k<<<dim3(NN, 24), 128, 0, stream>>>(S_hist, prm[P_tt_in_w], prm[P_tt_in_b],
      prm[P_tt_weekly], prm[P_tt_daily], prm[P_tt_pos], widx, didx, XG, Xtt, NN);

  // ---- temporal transformer: 2 fused layers over 5000 sequences of len 24 ----
  enc_layer_kernel<24, 32><<<NN, 256, 0, stream>>>(Xtt,
      t_enc[0][0], prm[P_tt0_bqkv], t_enc[0][1], prm[P_tt0_bo],
      prm[P_tt0_ln1g], prm[P_tt0_ln1b], t_enc[0][2], prm[P_tt0_b1],
      t_enc[0][3], prm[P_tt0_b2], prm[P_tt0_ln2g], prm[P_tt0_ln2b]);
  enc_layer_kernel<24, 32><<<NN, 256, 0, stream>>>(Xtt,
      t_enc[1][0], prm[P_tt1_bqkv], t_enc[1][1], prm[P_tt1_bo],
      prm[P_tt1_ln1g], prm[P_tt1_ln1b], t_enc[1][2], prm[P_tt1_b1],
      t_enc[1][3], prm[P_tt1_b2], prm[P_tt1_ln2g], prm[P_tt1_ln2b]);

  // ---- H_traf_time = x[:, -1, :] @ tt_out_w^T + b ----
  last_tok_k<<<(NNP * 128 + 255) / 256, 256, 0, stream>>>(Xtt, xlh, NN);
  gemm(xlh, t_ttout, prm[P_tt_out_b], Htt, NNP, 128, 128, 128);

  // ---- GAT #2 (trajectory, q=k=H0@tg_W^T) -> H_traj ----
  gemm(H0h, t_tgW, nullptr, qb, NNP, 128, 128, 128);   // qb reused as g
  gat_dots_k<<<(NN + 255) / 256, 256, 0, stream>>>(qb, qb, prm[P_tg_asrc], prm[P_tg_adst], ka, qa, NN);
  fill(smax, 0x00800000u, NNP);
  fill(ssum, 0u, NNP);
  fill(Htraj, 0u, NNP * 128);
  gat_logits_k<<<(EE + 255) / 256, 256, 0, stream>>>(esrc, edst, ka, qa, statc, 0, elog, smax, EE);
  gat_expsum_k<<<(EE + 255) / 256, 256, 0, stream>>>(edst, smax, elog, ssum, EE);
  gat_agg_k<<<(EE * 32 + 255) / 256, 256, 0, stream>>>(esrc, edst, elog, ssum, qb, Htraj, EE);

  // ---- trajectory tokens ----
  gather_h_k<<<BL, 128, 0, stream>>>(H0, Htt, Htraj, tnodes, Hgh);
  gemm(Hgh, t_tjnode, prm[P_tj_node_b], Gb, BL, 128, 128, 128);
  build_xt_k<<<dim3(64, 65), 128, 0, stream>>>(Gb, times, prm[P_tj_time_w],
                                               prm[P_tj_time_b], prm[P_tj_cls], xt);

  // ---- trajectory transformer: 2 fused layers over 64 sequences of len 65 ----
  enc_layer_kernel<65, 80><<<64, 256, 0, stream>>>(xt,
      t_enc[2][0], prm[P_tj0_bqkv], t_enc[2][1], prm[P_tj0_bo],
      prm[P_tj0_ln1g], prm[P_tj0_ln1b], t_enc[2][2], prm[P_tj0_b1],
      t_enc[2][3], prm[P_tj0_b2], prm[P_tj0_ln2g], prm[P_tj0_ln2b]);
  enc_layer_kernel<65, 80><<<64, 256, 0, stream>>>(xt,
      t_enc[3][0], prm[P_tj1_bqkv], t_enc[3][1], prm[P_tj1_bo],
      prm[P_tj1_ln1g], prm[P_tj1_ln1b], t_enc[3][2], prm[P_tj1_b1],
      t_enc[3][3], prm[P_tj1_b2], prm[P_tj1_ln2g], prm[P_tj1_ln2b]);

  // ---- final logits: [4096,128] @ [5000,128]^T + bias (padded ldb, branch-free) ----
  pos_gather_k<<<BL, 128, 0, stream>>>(xt, posh);
  gemm(posh, t_mtp, prm[P_mtp_seg_b], (float*)d_out, BL, 5000, MTP_LD, 5000);
}
